// DenoisingNetwork_44117904065181
// MI455X (gfx1250) — matrix operand order repaired and checked by reading, not compiled, not run on hardware
//
#include <hip/hip_runtime.h>

#define Hdim 256
#define H3   768
#define KMIX 20
#define NTt  10
#define ETt  4
#define MPREV 100
#define MPAD  112
#define NLAYERS 5

typedef _Float16 half_t;
typedef __attribute__((ext_vector_type(16))) _Float16 v16h;
typedef __attribute__((ext_vector_type(8)))  float    v8f;

union AFrag { v16h v; uint4 q[2]; };

__device__ __forceinline__ v8f wmma16(v16h a, v16h b, v8f c) {
  return __builtin_amdgcn_wmma_f32_16x16x32_f16(false, a, false, b, (short)0, c, false, false);
}

// A-matrix 16x32 f16 fragment (ISA layout: lanes 0-15 K={0..7,16..23}, lanes 16-31 K={8..15,24..31})
__device__ __forceinline__ v16h load_a(const half_t* base, int ld, int row0, int k0) {
  int lane = threadIdx.x & 31;
  const half_t* p = base + (size_t)(row0 + (lane & 15)) * ld + k0 + ((lane >> 4) * 8);
  AFrag f;
  f.q[0] = *(const uint4*)p;
  f.q[1] = *(const uint4*)(p + 16);
  return f.v;
}

__device__ __forceinline__ v16h load_a_clamp(const half_t* base, int ld, int row0, int k0, int maxrow) {
  int lane = threadIdx.x & 31;
  int r = row0 + (lane & 15);
  if (r > maxrow) r = maxrow;
  const half_t* p = base + (size_t)r * ld + k0 + ((lane >> 4) * 8);
  AFrag f;
  f.q[0] = *(const uint4*)p;
  f.q[1] = *(const uint4*)(p + 16);
  return f.v;
}

// B-matrix 32x16 f16 fragment from transposed weights Wt[N][K]: lane holds col n=lane&15,
// lanes 0-15 K={0..7,16..23}, lanes 16-31 K={8..15,24..31} (matches A).
__device__ __forceinline__ v16h load_b(const half_t* base, int ld, int col0, int k0) {
  int lane = threadIdx.x & 31;
  const half_t* p = base + (size_t)(col0 + (lane & 15)) * ld + k0 + ((lane >> 4) * 8);
  AFrag f;
  f.q[0] = *(const uint4*)p;
  f.q[1] = *(const uint4*)(p + 16);
  return f.v;
}

__device__ __forceinline__ float sigmoidf_(float x) { return 1.0f / (1.0f + expf(-x)); }

// ---------------------------------------------------------------- utility kernels
__global__ void zero_f32_kernel(float* __restrict__ p, size_t n) {
  for (size_t i = (size_t)blockIdx.x * 256 + threadIdx.x; i < n; i += (size_t)gridDim.x * 256)
    p[i] = 0.0f;
}

__global__ void cvt_f16_kernel(const float* __restrict__ in, half_t* __restrict__ out, size_t n) {
  for (size_t i = (size_t)blockIdx.x * 256 + threadIdx.x; i < n; i += (size_t)gridDim.x * 256)
    out[i] = (half_t)in[i];
}

// in: f32 row-major [batch][K][N]  ->  out: f16 [batch][N][K]
__global__ void transpose_f16_kernel(const float* __restrict__ in, half_t* __restrict__ out,
                                     int K, int N, int batch) {
  size_t total = (size_t)batch * K * N;
  for (size_t idx = (size_t)blockIdx.x * 256 + threadIdx.x; idx < total; idx += (size_t)gridDim.x * 256) {
    size_t kn = (size_t)K * N;
    size_t b = idx / kn;
    size_t r = idx - b * kn;
    int k = (int)(r / N);
    int n = (int)(r - (size_t)k * N);
    out[b * kn + (size_t)n * K + k] = (half_t)in[idx];
  }
}

__global__ void embed_nodes_kernel(const int* __restrict__ xn, const float* __restrict__ emb,
                                   float* __restrict__ h, half_t* __restrict__ h16, size_t n) {
  for (size_t i = (size_t)blockIdx.x * 256 + threadIdx.x; i < n; i += (size_t)gridDim.x * 256) {
    int node = (int)(i >> 8);
    int c = (int)(i & 255);
    float v = emb[(size_t)xn[node] * Hdim + c];
    h[i] = v;
    h16[i] = (half_t)v;
  }
}

// ---------------------------------------------------------------- edge message kernel
// feat[e] = [h[dst] | h[src] | eemb[attr]] gathered once in f16; two GEMMs (Wm1, Wa1) share
// the staged A operand. B tiles (32x256 per weight, 16KB each) are staged cooperatively
// into LDS each K-chunk so the WMMA stream is fed by ds_load_b128 instead of per-fragment
// global round-trips; the B-staging buffer aliases the Hid buffer (disjoint phases).
__device__ __forceinline__ void feat_gemm_dual(const half_t* __restrict__ h16,
                                               const half_t* __restrict__ etab,
                                               const int* Sdst, const int* Ssrc, const int* Sattr,
                                               half_t* As, half_t* Bst,
                                               const half_t* __restrict__ Wm1t,
                                               const half_t* __restrict__ Wa1t,
                                               int rowblk, int colbase,
                                               v8f acc1[8], v8f acc2[8]) {
  const int tid = threadIdx.x;
  const int e  = tid >> 2;
  const int kk = (tid & 3) * 8;
  half_t* Bm = Bst;                 // [256 cols][32 k]
  half_t* Ba = Bst + 32 * Hdim;     // second 16KB half
  for (int kc = 0; kc < 24; ++kc) {
    int k = kc * 32 + kk;
    const half_t* sp;
    if (k < 256)      sp = h16 + (size_t)Sdst[e] * Hdim + k;
    else if (k < 512) sp = h16 + (size_t)Ssrc[e] * Hdim + (k - 256);
    else              sp = etab + (size_t)Sattr[e] * Hdim + (k - 512);
    __syncthreads();                       // previous iteration done reading As/B stages
    // stage A gather (16B per thread)
    *(uint4*)(As + e * 32 + kk) = *(const uint4*)sp;
    // stage both B tiles: thread t copies the 32-half K-slice of weight column t (64B each)
    {
      const uint4* s1 = (const uint4*)(Wm1t + (size_t)tid * H3 + kc * 32);
      const uint4* s2 = (const uint4*)(Wa1t + (size_t)tid * H3 + kc * 32);
      uint4* d1 = (uint4*)(Bm + tid * 32);
      uint4* d2 = (uint4*)(Ba + tid * 32);
      d1[0] = s1[0]; d1[1] = s1[1]; d1[2] = s1[2]; d1[3] = s1[3];
      d2[0] = s2[0]; d2[1] = s2[1]; d2[2] = s2[2]; d2[3] = s2[3];
    }
    // prefetch next K-chunk of the weight streams into cache
    if (kc < 23) {
      __builtin_prefetch(Wm1t + (size_t)tid * H3 + (kc + 1) * 32, 0, 1);
      __builtin_prefetch(Wa1t + (size_t)tid * H3 + (kc + 1) * 32, 0, 1);
    }
    __syncthreads();
    v16h a = load_a(As, 32, rowblk * 16, 0);
#pragma unroll
    for (int g = 0; g < 2; ++g) {
      v16h bm[4], ba[4];
#pragma unroll
      for (int i = 0; i < 4; ++i) {
        int col = colbase + 16 * (g * 4 + i);
        bm[i] = load_b(Bm, 32, col, 0);
        ba[i] = load_b(Ba, 32, col, 0);
      }
#pragma unroll
      for (int i = 0; i < 4; ++i) {
        acc1[g * 4 + i] = wmma16(a, bm[i], acc1[g * 4 + i]);
        acc2[g * 4 + i] = wmma16(a, ba[i], acc2[g * 4 + i]);
      }
    }
  }
  __syncthreads();
}

__global__ __launch_bounds__(256) void edge_msg_kernel(
    const half_t* __restrict__ h16,
    const int* __restrict__ eidx, const int* __restrict__ eattr,
    const half_t* __restrict__ etab,
    const half_t* __restrict__ Wm1t, const float* __restrict__ bm1,
    const half_t* __restrict__ Wm2t, const float* __restrict__ bm2,
    const half_t* __restrict__ Wa1t, const float* __restrict__ ba1,
    const float* __restrict__ Wa2,  const float* __restrict__ ba2,
    float* __restrict__ agg, int E) {
  __shared__ __align__(16) half_t As[64 * 32];
  __shared__ __align__(16) half_t BigB[64 * Hdim];   // B staging during GEMM, Hid afterwards
  __shared__ float Att[64];
  __shared__ int Sdst[64], Ssrc[64], Sattr[64];

  const int tid = threadIdx.x;
  const int lane = tid & 31;
  const int wave = tid >> 5;
  const int rowblk = wave & 3;
  const int colbase = (wave >> 2) * 128;
  const int e0 = blockIdx.x * 64;
  half_t* Hid = BigB;

  if (tid < 64) {
    int eg = e0 + tid;
    int ec = eg < E ? eg : E - 1;
    Ssrc[tid]  = eidx[ec];        // edge_index[0] = source (x_j)
    Sdst[tid]  = eidx[E + ec];    // edge_index[1] = dest   (x_i)
    Sattr[tid] = eattr[ec];
    Att[tid] = 0.0f;
  }
  __syncthreads();

  v8f zero = {};
  v8f acc1[8], acc2[8];
  const int mb = rowblk * 16 + (lane >> 4) * 8;

  // fused pass 1+2: hidden (Wm1) and attention-hidden (Wa1) off one staged A operand
  for (int j = 0; j < 8; ++j) { acc1[j] = zero; acc2[j] = zero; }
  feat_gemm_dual(h16, etab, Sdst, Ssrc, Sattr, As, BigB, Wm1t, Wa1t, rowblk, colbase, acc1, acc2);

  // epilogue 2 first (BigB is about to be overwritten by Hid):
  // attention logit = relu(. + ba1) . Wa2, reduced per edge via LDS atomics
  {
    float part[8];
#pragma unroll
    for (int v = 0; v < 8; ++v) part[v] = 0.0f;
#pragma unroll
    for (int j = 0; j < 8; ++j) {
      int n = colbase + 16 * j + (lane & 15);
      float ba = ba1[n];
      float w = Wa2[n];
#pragma unroll
      for (int v = 0; v < 8; ++v) {
        float x = acc2[j][v] + ba;
        part[v] += (x > 0.0f ? x : 0.0f) * w;
      }
    }
#pragma unroll
    for (int v = 0; v < 8; ++v) atomicAdd(&Att[mb + v], part[v]);
  }

  // epilogue 1: Hid = relu(. + bm1) stored f16 into BigB (all reads of B stage are done)
#pragma unroll
  for (int j = 0; j < 8; ++j) {
    int n = colbase + 16 * j + (lane & 15);
    float bj = bm1[n];
#pragma unroll
    for (int v = 0; v < 8; ++v) {
      float x = acc1[j][v] + bj;
      Hid[(mb + v) * Hdim + n] = (half_t)(x > 0.0f ? x : 0.0f);
    }
  }
  __syncthreads();
  if (tid < 64) Att[tid] = sigmoidf_(Att[tid] + ba2[0]);
  __syncthreads();

  // pass 3: msg = Hid @ Wm2 + bm2 ; agg[dst] += msg * att
#pragma unroll
  for (int j = 0; j < 8; ++j) acc1[j] = zero;
  for (int kc = 0; kc < 8; ++kc) {
    v16h a = load_a(Hid, Hdim, rowblk * 16, kc * 32);
#pragma unroll
    for (int g = 0; g < 2; ++g) {
      v16h bw[4];
#pragma unroll
      for (int i = 0; i < 4; ++i)
        bw[i] = load_b(Wm2t, Hdim, colbase + 16 * (g * 4 + i), kc * 32);
#pragma unroll
      for (int i = 0; i < 4; ++i)
        acc1[g * 4 + i] = wmma16(a, bw[i], acc1[g * 4 + i]);
    }
  }
#pragma unroll
  for (int j = 0; j < 8; ++j) {
    int n = colbase + 16 * j + (lane & 15);
    float bj = bm2[n];
#pragma unroll
    for (int v = 0; v < 8; ++v) {
      int m = mb + v;
      if (e0 + m < E) {
        float val = (acc1[j][v] + bj) * Att[m];
        unsafeAtomicAdd(&agg[(size_t)Sdst[m] * Hdim + n], val);
      }
    }
  }
}

// ---------------------------------------------------------------- GRU kernel (fused gi/gh WMMA)
__global__ __launch_bounds__(256) void gru_kernel(
    const half_t* __restrict__ agg16, const half_t* __restrict__ h16,
    const float* __restrict__ hold,
    const half_t* __restrict__ Wiht, const float* __restrict__ bih,
    const half_t* __restrict__ Whht, const float* __restrict__ bhh,
    float* __restrict__ hnew, half_t* __restrict__ hnew16, int N) {
  const int tid = threadIdx.x;
  const int lane = tid & 31;
  const int wave = tid >> 5;
  const int rowblk = wave & 3;
  const int colhalf = wave >> 2;
  const int n0 = blockIdx.x * 64;
  const int maxrow = N - 1;

  v8f zero = {};
  for (int hp = 0; hp < 2; ++hp) {
    const int cb = colhalf * 128 + hp * 64;
    v8f accR[4], accZ[4], accI[4], accHN[4];
#pragma unroll
    for (int t = 0; t < 4; ++t) { accR[t] = zero; accZ[t] = zero; accI[t] = zero; accHN[t] = zero; }
    for (int kc = 0; kc < 8; ++kc) {
      v16h aA = load_a_clamp(agg16, Hdim, n0 + rowblk * 16, kc * 32, maxrow);
      v16h aH = load_a_clamp(h16,   Hdim, n0 + rowblk * 16, kc * 32, maxrow);
#pragma unroll
      for (int t = 0; t < 4; ++t) {
        int ct = cb + 16 * t;
        v16h bw[6];
        bw[0] = load_b(Wiht, Hdim, ct,       kc * 32);
        bw[1] = load_b(Whht, Hdim, ct,       kc * 32);
        bw[2] = load_b(Wiht, Hdim, 256 + ct, kc * 32);
        bw[3] = load_b(Whht, Hdim, 256 + ct, kc * 32);
        bw[4] = load_b(Wiht, Hdim, 512 + ct, kc * 32);
        bw[5] = load_b(Whht, Hdim, 512 + ct, kc * 32);
        accR[t]  = wmma16(aA, bw[0], accR[t]);
        accR[t]  = wmma16(aH, bw[1], accR[t]);
        accZ[t]  = wmma16(aA, bw[2], accZ[t]);
        accZ[t]  = wmma16(aH, bw[3], accZ[t]);
        accI[t]  = wmma16(aA, bw[4], accI[t]);
        accHN[t] = wmma16(aH, bw[5], accHN[t]);
      }
    }
    const int mb = rowblk * 16 + (lane >> 4) * 8;
#pragma unroll
    for (int t = 0; t < 4; ++t) {
      int n = cb + 16 * t + (lane & 15);
      float bir = bih[n],       bhr = bhh[n];
      float biz = bih[256 + n], bhz = bhh[256 + n];
      float bin = bih[512 + n], bhn = bhh[512 + n];
#pragma unroll
      for (int v = 0; v < 8; ++v) {
        int node = n0 + mb + v;
        if (node < N) {
          float r  = sigmoidf_(accR[t][v] + bir + bhr);
          float z  = sigmoidf_(accZ[t][v] + biz + bhz);
          float nn = tanhf(accI[t][v] + bin + r * (accHN[t][v] + bhn));
          float ho = hold[(size_t)node * Hdim + n];
          float out = (1.0f - z) * nn + z * ho;
          hnew[(size_t)node * Hdim + n] = out;
          hnew16[(size_t)node * Hdim + n] = (half_t)out;
        }
      }
    }
  }
}

// ---------------------------------------------------------------- pooling / heads
__global__ void colmean_kernel(const float* __restrict__ h, float* __restrict__ hG, int N) {
  __shared__ float red[256];
  int c = blockIdx.x;
  float s = 0.0f;
  for (int n = threadIdx.x; n < N; n += 256) s += h[(size_t)n * Hdim + c];
  red[threadIdx.x] = s;
  __syncthreads();
  for (int o = 128; o > 0; o >>= 1) {
    if (threadIdx.x < o) red[threadIdx.x] += red[threadIdx.x + o];
    __syncthreads();
  }
  if (threadIdx.x == 0) hG[c] = red[0] / (float)N;
}

__global__ void node_head_kernel(const float* __restrict__ h, const float* __restrict__ hG,
                                 const int* __restrict__ tgtp,
                                 const float* __restrict__ W1, const float* __restrict__ b1,
                                 const float* __restrict__ W2, const float* __restrict__ b2,
                                 float* __restrict__ out) {
  __shared__ float hid[Hdim];
  __shared__ float lg[16];
  int tgt = tgtp[0];
  int i = threadIdx.x;
  float s = b1[i];
  for (int c = 0; c < Hdim; ++c) s += hG[c] * W1[c * Hdim + i];
  for (int c = 0; c < Hdim; ++c) s += h[(size_t)tgt * Hdim + c] * W1[(Hdim + c) * Hdim + i];
  hid[i] = s > 0.0f ? s : 0.0f;
  __syncthreads();
  if (i < NTt) {
    float t = b2[i];
    for (int j = 0; j < Hdim; ++j) t += hid[j] * W2[j * NTt + i];
    lg[i] = t;
  }
  __syncthreads();
  if (i == 0) {
    float mx = lg[0];
    for (int q = 1; q < NTt; ++q) mx = fmaxf(mx, lg[q]);
    float e[NTt]; float sum = 0.0f;
    for (int q = 0; q < NTt; ++q) { e[q] = expf(lg[q] - mx); sum += e[q]; }
    for (int q = 0; q < NTt; ++q) out[q] = e[q] / sum;
  }
}

__global__ void build_ei_kernel(const float* __restrict__ h, const float* __restrict__ hG,
                                const int* __restrict__ tgtp, const int* __restrict__ prev,
                                half_t* __restrict__ ei16) {
  int m = blockIdx.x;
  int c = threadIdx.x;
  int tgt = tgtp[0];
  float a = 0.0f, b = 0.0f, d = 0.0f;
  if (m < MPREV) {
    a = hG[c];
    b = h[(size_t)tgt * Hdim + c];
    d = h[(size_t)prev[m] * Hdim + c];
  }
  ei16[(size_t)m * H3 + c]        = (half_t)a;
  ei16[(size_t)m * H3 + 256 + c]  = (half_t)b;
  ei16[(size_t)m * H3 + 512 + c]  = (half_t)d;
}

// 112x768 @ 768x256 WMMA GEMM + relu into LDS (shared by mix/edge heads)
__device__ __forceinline__ void head_gemm(const half_t* __restrict__ ei16,
                                          const half_t* __restrict__ W1t,
                                          const float* __restrict__ b1, half_t* Hid) {
  const int lane = threadIdx.x & 31;
  const int wave = threadIdx.x >> 5;
  v8f zero = {};
  v8f acc[14];
#pragma unroll
  for (int i = 0; i < 14; ++i) acc[i] = zero;
  for (int kc = 0; kc < 24; ++kc) {
#pragma unroll
    for (int g = 0; g < 2; ++g) {
      v16h av[7], bv[7];
#pragma unroll
      for (int i = 0; i < 7; ++i) {
        int t = wave * 14 + g * 7 + i;
        int rb = t >> 4, ct = t & 15;
        av[i] = load_a(ei16, H3, rb * 16, kc * 32);
        bv[i] = load_b(W1t, H3, ct * 16, kc * 32);
      }
#pragma unroll
      for (int i = 0; i < 7; ++i)
        acc[g * 7 + i] = wmma16(av[i], bv[i], acc[g * 7 + i]);
    }
  }
  const int mbof = (lane >> 4) * 8;
#pragma unroll
  for (int i = 0; i < 14; ++i) {
    int t = wave * 14 + i;
    int rb = t >> 4, ct = t & 15;
    int n = ct * 16 + (lane & 15);
    float bb = b1[n];
#pragma unroll
    for (int v = 0; v < 8; ++v) {
      float x = acc[i][v] + bb;
      Hid[(rb * 16 + mbof + v) * Hdim + n] = (half_t)(x > 0.0f ? x : 0.0f);
    }
  }
  __syncthreads();
}

__global__ __launch_bounds__(256) void mix_head_kernel(
    const half_t* __restrict__ ei16, const half_t* __restrict__ W1t,
    const float* __restrict__ b1, const float* __restrict__ W2,
    const float* __restrict__ b2, float* __restrict__ mixout) {
  __shared__ __align__(16) half_t Hid[MPAD * Hdim];
  head_gemm(ei16, W1t, b1, Hid);
  int m = threadIdx.x;
  if (m < MPREV) {
    float lg[KMIX];
    for (int q = 0; q < KMIX; ++q) {
      float s = b2[q];
      for (int i = 0; i < Hdim; ++i) s += (float)Hid[m * Hdim + i] * W2[i * KMIX + q];
      lg[q] = s;
    }
    float mx = lg[0];
    for (int q = 1; q < KMIX; ++q) mx = fmaxf(mx, lg[q]);
    float sum = 0.0f;
    for (int q = 0; q < KMIX; ++q) { lg[q] = expf(lg[q] - mx); sum += lg[q]; }
    for (int q = 0; q < KMIX; ++q) mixout[m * KMIX + q] = lg[q] / sum;
  }
}

__global__ __launch_bounds__(256) void edge_head_kernel(
    const half_t* __restrict__ ei16, const half_t* __restrict__ W1tAll,
    const float* __restrict__ b1All, const float* __restrict__ W2All,
    const float* __restrict__ b2All, const float* __restrict__ mixout,
    float* __restrict__ out) {
  int k = blockIdx.x;
  __shared__ __align__(16) half_t Hid[MPAD * Hdim];
  head_gemm(ei16, W1tAll + (size_t)k * H3 * Hdim, b1All + k * Hdim, Hid);
  int m = threadIdx.x;
  if (m < MPREV) {
    float lg[ETt];
    for (int e = 0; e < ETt; ++e) {
      float s = b2All[k * ETt + e];
      for (int i = 0; i < Hdim; ++i)
        s += (float)Hid[m * Hdim + i] * W2All[(size_t)k * Hdim * ETt + i * ETt + e];
      lg[e] = s;
    }
    float mx = fmaxf(fmaxf(lg[0], lg[1]), fmaxf(lg[2], lg[3]));
    float sum = 0.0f;
    for (int e = 0; e < ETt; ++e) { lg[e] = expf(lg[e] - mx); sum += lg[e]; }
    float w = mixout[m * KMIX + k];
    for (int e = 0; e < ETt; ++e)
      unsafeAtomicAdd(&out[NTt + m * ETt + e], w * lg[e] / sum);
  }
}

// ---------------------------------------------------------------- host launcher
extern "C" void kernel_launch(void* const* d_in, const int* in_sizes, int n_in,
                              void* d_out, int out_size, void* d_ws, size_t ws_size,
                              hipStream_t stream) {
  const int N = in_sizes[0];
  const int E = in_sizes[2];

  const int* x_nodes = (const int*)d_in[0];
  const int* eidx    = (const int*)d_in[1];
  const int* eattr   = (const int*)d_in[2];
  const int* tgtp    = (const int*)d_in[3];
  const int* prev    = (const int*)d_in[4];
  const float* node_emb = (const float*)d_in[5];
  const float* etabf    = (const float*)d_in[6];

  char* ws = (char*)d_ws;
  size_t off = 0;
  auto alloc = [&](size_t bytes) -> void* {
    void* p = (void*)(ws + off);
    off += (bytes + 255) & ~(size_t)255;
    return p;
  };

  float*  hA    = (float*)alloc((size_t)N * Hdim * 4);
  float*  hB    = (float*)alloc((size_t)N * Hdim * 4);
  half_t* h16A  = (half_t*)alloc((size_t)N * Hdim * 2);
  half_t* h16B  = (half_t*)alloc((size_t)N * Hdim * 2);
  float*  agg   = (float*)alloc((size_t)N * Hdim * 4);
  half_t* agg16 = (half_t*)alloc((size_t)N * Hdim * 2);
  half_t* etab16 = (half_t*)alloc((size_t)ETt * Hdim * 2);
  half_t *Wm1t[NLAYERS], *Wa1t[NLAYERS], *Wm2t[NLAYERS], *Wiht[NLAYERS], *Whht[NLAYERS];
  for (int l = 0; l < NLAYERS; ++l) {
    Wm1t[l] = (half_t*)alloc((size_t)H3 * Hdim * 2);
    Wa1t[l] = (half_t*)alloc((size_t)H3 * Hdim * 2);
    Wm2t[l] = (half_t*)alloc((size_t)Hdim * Hdim * 2);
    Wiht[l] = (half_t*)alloc((size_t)H3 * Hdim * 2);
    Whht[l] = (half_t*)alloc((size_t)H3 * Hdim * 2);
  }
  half_t* mixW1t  = (half_t*)alloc((size_t)H3 * Hdim * 2);
  half_t* edgeW1t = (half_t*)alloc((size_t)KMIX * H3 * Hdim * 2);
  float*  mixout  = (float*)alloc((size_t)MPREV * KMIX * 4);
  half_t* ei16    = (half_t*)alloc((size_t)MPAD * H3 * 2);
  float*  hG      = (float*)alloc(Hdim * 4);

  // zero output (edge part is atomically accumulated)
  zero_f32_kernel<<<4, 256, 0, stream>>>((float*)d_out, (size_t)out_size);

  // weight prep: f32 [K][N] -> f16 [N][K]
  for (int l = 0; l < NLAYERS; ++l) {
    const float* Wm1 = (const float*)d_in[7 + 12 * l + 0];
    const float* Wm2 = (const float*)d_in[7 + 12 * l + 2];
    const float* Wa1 = (const float*)d_in[7 + 12 * l + 4];
    const float* Wih = (const float*)d_in[7 + 12 * l + 8];
    const float* Whh = (const float*)d_in[7 + 12 * l + 10];
    int tb = (H3 * Hdim + 255) / 256;
    transpose_f16_kernel<<<tb, 256, 0, stream>>>(Wm1, Wm1t[l], H3, Hdim, 1);
    transpose_f16_kernel<<<tb, 256, 0, stream>>>(Wa1, Wa1t[l], H3, Hdim, 1);
    transpose_f16_kernel<<<(Hdim * Hdim + 255) / 256, 256, 0, stream>>>(Wm2, Wm2t[l], Hdim, Hdim, 1);
    transpose_f16_kernel<<<tb, 256, 0, stream>>>(Wih, Wiht[l], Hdim, H3, 1);
    transpose_f16_kernel<<<tb, 256, 0, stream>>>(Whh, Whht[l], Hdim, H3, 1);
  }
  transpose_f16_kernel<<<(H3 * Hdim + 255) / 256, 256, 0, stream>>>(
      (const float*)d_in[75], mixW1t, H3, Hdim, 1);
  transpose_f16_kernel<<<(int)(((size_t)KMIX * H3 * Hdim + 255) / 256), 256, 0, stream>>>(
      (const float*)d_in[71], edgeW1t, H3, Hdim, KMIX);
  cvt_f16_kernel<<<(ETt * Hdim + 255) / 256, 256, 0, stream>>>(etabf, etab16, (size_t)ETt * Hdim);

  // initial embeddings
  embed_nodes_kernel<<<(int)(((size_t)N * Hdim + 255) / 256), 256, 0, stream>>>(
      x_nodes, node_emb, hA, h16A, (size_t)N * Hdim);

  float*  hc = hA;   float*  hn = hB;
  half_t* h16c = h16A; half_t* h16n = h16B;
  for (int l = 0; l < NLAYERS; ++l) {
    const float* bm1 = (const float*)d_in[7 + 12 * l + 1];
    const float* bm2 = (const float*)d_in[7 + 12 * l + 3];
    const float* ba1 = (const float*)d_in[7 + 12 * l + 5];
    const float* Wa2 = (const float*)d_in[7 + 12 * l + 6];
    const float* ba2 = (const float*)d_in[7 + 12 * l + 7];
    const float* bih = (const float*)d_in[7 + 12 * l + 9];
    const float* bhh = (const float*)d_in[7 + 12 * l + 11];

    zero_f32_kernel<<<(int)(((size_t)N * Hdim + 255) / 256), 256, 0, stream>>>(agg, (size_t)N * Hdim);
    edge_msg_kernel<<<(E + 63) / 64, 256, 0, stream>>>(
        h16c, eidx, eattr, etab16, Wm1t[l], bm1, Wm2t[l], bm2, Wa1t[l], ba1, Wa2, ba2, agg, E);
    cvt_f16_kernel<<<(int)(((size_t)N * Hdim + 255) / 256), 256, 0, stream>>>(
        agg, agg16, (size_t)N * Hdim);
    gru_kernel<<<(N + 63) / 64, 256, 0, stream>>>(
        agg16, h16c, hc, Wiht[l], bih, Whht[l], bhh, hn, h16n, N);
    float* tf = hc; hc = hn; hn = tf;
    half_t* th = h16c; h16c = h16n; h16n = th;
  }

  // heads
  colmean_kernel<<<Hdim, 256, 0, stream>>>(hc, hG, N);
  node_head_kernel<<<1, 256, 0, stream>>>(
      hc, hG, tgtp,
      (const float*)d_in[67], (const float*)d_in[68],
      (const float*)d_in[69], (const float*)d_in[70], (float*)d_out);
  build_ei_kernel<<<MPAD, 256, 0, stream>>>(hc, hG, tgtp, prev, ei16);
  mix_head_kernel<<<1, 256, 0, stream>>>(
      ei16, mixW1t, (const float*)d_in[76], (const float*)d_in[77],
      (const float*)d_in[78], mixout);
  edge_head_kernel<<<KMIX, 256, 0, stream>>>(
      ei16, edgeW1t, (const float*)d_in[72], (const float*)d_in[73],
      (const float*)d_in[74], mixout, (float*)d_out);
}
